// Attention_10136122818776
// MI455X (gfx1250) — matrix operand order repaired and checked by reading, not compiled, not run on hardware
//
#include <hip/hip_runtime.h>
#include <cmath>
#include <cstdint>

#define SEQ   2048
#define DIM   1024
#define HEADS 8
#define DHEAD 64
#define INNER 512
#define BATCH 2
#define MROWS (BATCH * SEQ)   // 4096

typedef __attribute__((ext_vector_type(16))) __bf16 v16bf;
typedef __attribute__((ext_vector_type(8)))  float  v8f;
typedef unsigned int u32x4 __attribute__((ext_vector_type(4)));
typedef int          i32x8 __attribute__((ext_vector_type(8)));
typedef int          i32x4 __attribute__((ext_vector_type(4)));

union FragAB {
    v16bf          v;
    unsigned       u32[8];
    unsigned short u16[16];
};

__device__ __forceinline__ unsigned short f2bf(float f) {
    unsigned u = __float_as_uint(f);
    u += 0x7FFFu + ((u >> 16) & 1u);          // round-to-nearest-even
    return (unsigned short)(u >> 16);
}

// ---------------------------------------------------------------------------
// TDM: DMA a 2-D bf16 tile [tileN rows x tileK elts] (row stride = strideElts)
// from global into LDS at lds_off. D# per CDNA5 ISA.
// 6-arg builtin form (amdgpu-toolchain clang-23).
// ---------------------------------------------------------------------------
__device__ __forceinline__ void tdm_load_tile(const unsigned short* gsrc, unsigned lds_off,
                                              unsigned tileK, unsigned tileN,
                                              unsigned tensorK, unsigned strideElts) {
    unsigned long long ga = (unsigned long long)(uintptr_t)gsrc;
    // group0: count=1 | lds_addr | global_addr[56:0] | type=2 ("image")
    u32x4 g0 = { 1u,
                 lds_off,
                 (unsigned)ga,
                 (unsigned)((ga >> 32) & 0x01FFFFFFu) | 0x80000000u };
    // group1: mask=0, data_size=1 (2B), tensor_dim0, tensor_dim1, tile dims, strides
    unsigned td0 = tensorK, td1 = tileN, s0 = strideElts;
    i32x8 g1 = { (int)(1u << 16),                               // data_size=1 (2 bytes)
                 (int)(td0 << 16),                              // tensor_dim0[15:0]
                 (int)((td0 >> 16) | (td1 << 16)),              // td0[31:16] | td1[15:0]
                 (int)((td1 >> 16) | (tileK << 16)),            // td1[31:16] | tile_dim0
                 (int)(tileN & 0xFFFFu),                        // tile_dim1 (tile_dim2=0)
                 (int)s0,                                       // tensor_dim0_stride[31:0]
                 0,                                             // stride0 hi | stride1 lo
                 0 };
    i32x4 z4 = { 0, 0, 0, 0 };
    i32x8 z8 = { 0, 0, 0, 0, 0, 0, 0, 0 };
    __builtin_amdgcn_tensor_load_to_lds(g0, g1, z4, z4, z8, 0);
}

// ---------------------------------------------------------------------------
// Conversions
// ---------------------------------------------------------------------------
__global__ void k_cvt_x(const float* __restrict__ x, unsigned short* __restrict__ xb, int n) {
    int i = blockIdx.x * blockDim.x + threadIdx.x;
    if (i < n) xb[i] = f2bf(x[i]);
}

// Wt[n*K + k] = W[k*ld + colOff + n]   (bf16, transposed so GEMM B-frags are K-contiguous)
__global__ void k_cvt_wT(const float* __restrict__ W, unsigned short* __restrict__ Wt,
                         int K, int N, int ld, int colOff) {
    int idx = blockIdx.x * blockDim.x + threadIdx.x;
    if (idx >= K * N) return;
    int n = idx / K, k = idx % K;
    Wt[idx] = f2bf(W[(size_t)k * ld + colOff + n]);
}

// T5 relative-position bias LUT: lut[h*SEQ + dist] = rel_emb[bucket(dist)][h] * sqrt(d)
__global__ void k_bias_lut(const float* __restrict__ rel_emb, float* __restrict__ lut) {
    int idx = blockIdx.x * blockDim.x + threadIdx.x;
    if (idx >= HEADS * SEQ) return;
    int h = idx >> 11, dist = idx & (SEQ - 1);
    int bucket;
    if (dist < 16) {
        bucket = dist;
    } else {
        int vb = 16 + (int)(logf((float)dist * (1.0f / 16.0f)) * (16.0f / logf(8.0f)));
        bucket = vb < 31 ? vb : 31;
    }
    lut[idx] = rel_emb[bucket * HEADS + h] * 8.0f;   // bias_scale = sqrt(64)
}

// ---------------------------------------------------------------------------
// bf16 WMMA GEMM: C_bf16[M,N] = scale * (A[M,K] @ Bt[N,K]^T)
// Block = 4 waves sharing one 64-col N-strip; B tile staged in LDS by the TDM
// (double-buffered), A streamed from global. vmode=1 stores V transposed.
// ---------------------------------------------------------------------------
__global__ __launch_bounds__(128)
void k_gemm_bf16(const unsigned short* __restrict__ A, const unsigned short* __restrict__ Bt,
                 unsigned short* __restrict__ C, int M, int N, int K, float scale, int vmode) {
    __shared__ unsigned short Bs[2][64 * 32];
    const int lane = threadIdx.x & 31;
    const int w    = threadIdx.x >> 5;
    const int nloc = lane & 15, g = lane >> 4;
    const int mB4  = (M >> 4) >> 2;                // blocks along M
    const int mBlk = blockIdx.x % mB4, nS = blockIdx.x / mB4;
    const int m0 = (mBlk * 4 + w) << 4, n0 = nS << 6;

    if (w == 0) {
        tdm_load_tile(Bt + (size_t)n0 * K, (unsigned)(uintptr_t)&Bs[0][0], 32, 64, K, K);
        __builtin_amdgcn_s_wait_tensorcnt(0);
    }
    __syncthreads();

    v8f c[4] = {};
    const unsigned short* arow = A + (size_t)(m0 + nloc) * K;
    int buf = 0;

    for (int k0 = 0; k0 < K; k0 += 32) {
        if (w == 0 && k0 + 32 < K)
            tdm_load_tile(Bt + (size_t)n0 * K + k0 + 32,
                          (unsigned)(uintptr_t)&Bs[buf ^ 1][0], 32, 64, K, K);
        FragAB a;
#pragma unroll
        for (int r = 0; r < 8; ++r) {
            int kk = k0 + (g << 4) + (r << 1);
            a.u32[r] = *(const unsigned*)(arow + kk);
        }
        __builtin_prefetch(arow + k0 + 64, 0, 1);
#pragma unroll
        for (int nb = 0; nb < 4; ++nb) {
            const unsigned short* brow = &Bs[buf][(nb * 16 + nloc) * 32 + (g << 4)];
            FragAB b;
#pragma unroll
            for (int r = 0; r < 8; ++r) b.u32[r] = *(const unsigned*)(brow + 2 * r);
            c[nb] = __builtin_amdgcn_wmma_f32_16x16x32_bf16(false, a.v, false, b.v,
                                                            (short)0, c[nb], false, false);
        }
        if (w == 0) __builtin_amdgcn_s_wait_tensorcnt(0);
        __syncthreads();
        buf ^= 1;
    }
#pragma unroll
    for (int nb = 0; nb < 4; ++nb) {
#pragma unroll
        for (int r = 0; r < 8; ++r) {
            int m   = m0 + (g << 3) + r;
            int col = n0 + nb * 16 + nloc;
            float val = c[nb][r] * scale;
            size_t addr;
            if (vmode) {
                int bidx = m >> 11, i = m & (SEQ - 1);
                int h = col >> 6, d = col & 63;
                addr = ((size_t)((bidx * HEADS + h) * DHEAD + d) << 11) + i;
            } else {
                addr = (size_t)m * N + col;
            }
            C[addr] = f2bf(val);
        }
    }
}

// Output GEMM: f32 C[M,N] = A[M,K] @ Bt[N,K]^T + bias[N]  (same TDM-staged structure)
__global__ __launch_bounds__(128)
void k_gemm_out(const unsigned short* __restrict__ A, const unsigned short* __restrict__ Bt,
                const float* __restrict__ bias, float* __restrict__ C, int M, int N, int K) {
    __shared__ unsigned short Bs[2][64 * 32];
    const int lane = threadIdx.x & 31;
    const int w    = threadIdx.x >> 5;
    const int nloc = lane & 15, g = lane >> 4;
    const int mB4  = (M >> 4) >> 2;
    const int mBlk = blockIdx.x % mB4, nS = blockIdx.x / mB4;
    const int m0 = (mBlk * 4 + w) << 4, n0 = nS << 6;

    if (w == 0) {
        tdm_load_tile(Bt + (size_t)n0 * K, (unsigned)(uintptr_t)&Bs[0][0], 32, 64, K, K);
        __builtin_amdgcn_s_wait_tensorcnt(0);
    }
    __syncthreads();

    v8f c[4] = {};
    const unsigned short* arow = A + (size_t)(m0 + nloc) * K;
    int buf = 0;

    for (int k0 = 0; k0 < K; k0 += 32) {
        if (w == 0 && k0 + 32 < K)
            tdm_load_tile(Bt + (size_t)n0 * K + k0 + 32,
                          (unsigned)(uintptr_t)&Bs[buf ^ 1][0], 32, 64, K, K);
        FragAB a;
#pragma unroll
        for (int r = 0; r < 8; ++r) {
            int kk = k0 + (g << 4) + (r << 1);
            a.u32[r] = *(const unsigned*)(arow + kk);
        }
        __builtin_prefetch(arow + k0 + 64, 0, 1);
#pragma unroll
        for (int nb = 0; nb < 4; ++nb) {
            const unsigned short* brow = &Bs[buf][(nb * 16 + nloc) * 32 + (g << 4)];
            FragAB b;
#pragma unroll
            for (int r = 0; r < 8; ++r) b.u32[r] = *(const unsigned*)(brow + 2 * r);
            c[nb] = __builtin_amdgcn_wmma_f32_16x16x32_bf16(false, a.v, false, b.v,
                                                            (short)0, c[nb], false, false);
        }
        if (w == 0) __builtin_amdgcn_s_wait_tensorcnt(0);
        __syncthreads();
        buf ^= 1;
    }
#pragma unroll
    for (int nb = 0; nb < 4; ++nb) {
#pragma unroll
        for (int r = 0; r < 8; ++r) {
            int m   = m0 + (g << 3) + r;
            int col = n0 + nb * 16 + nloc;
            C[(size_t)m * N + col] = c[nb][r] + bias[col];
        }
    }
}

// ---------------------------------------------------------------------------
// Causal flash attention with T5 bias.
// Wave = one 16-query tile; block = 4 waves; grid = (SEQ/64, B*H).
// Q pre-scaled by 1/sqrt(d). K is [b*n, h*d] bf16; V is transposed [b*h,64,2048].
// ---------------------------------------------------------------------------
__global__ __launch_bounds__(128)
void k_flash(const unsigned short* __restrict__ Q, const unsigned short* __restrict__ Km,
             const unsigned short* __restrict__ Vt, const float* __restrict__ lut,
             unsigned short* __restrict__ O) {
    __shared__ unsigned short sm[4 * 16 * 32];   // per-wave P staging (C-layout -> A-layout)
    const int lane = threadIdx.x & 31;
    const int w    = threadIdx.x >> 5;
    const int nloc = lane & 15, g = lane >> 4;
    const int bh = blockIdx.y, b = bh >> 3, h = bh & 7;
    const int i0 = blockIdx.x * 64 + w * 16;
    unsigned short* smw = sm + w * 512;
    const float* lutH = lut + h * SEQ;

    // Q A-fragments for K-dim d=0..31 and d=32..63
    FragAB aq0, aq1;
    {
        const unsigned short* qrow = Q + (size_t)(b * SEQ + i0 + nloc) * INNER + h * DHEAD;
#pragma unroll
        for (int r = 0; r < 8; ++r) {
            int kk = (g << 4) + (r << 1);
            aq0.u32[r] = *(const unsigned*)(qrow + kk);
            aq1.u32[r] = *(const unsigned*)(qrow + 32 + kk);
        }
    }

    v8f  o[4] = {};
    float m_run[8], l_run[8];
#pragma unroll
    for (int r = 0; r < 8; ++r) { m_run[r] = -3.0e38f; l_run[r] = 0.0f; }

    const int kbMax = (i0 + 15) >> 5;
    for (int kb = 0; kb <= kbMax; ++kb) {
        const int j0 = kb << 5;
        v8f s[2];
        // S = (Q/sqrt(d)) K^T for two 16-key halves (K-dim = 64 -> 2 WMMA each)
#pragma unroll
        for (int t = 0; t < 2; ++t) {
            const unsigned short* krow =
                Km + (size_t)(b * SEQ + j0 + t * 16 + nloc) * INNER + h * DHEAD + (g << 4);
            FragAB bk0, bk1;
#pragma unroll
            for (int r = 0; r < 8; ++r) {
                bk0.u32[r] = *(const unsigned*)(krow + 2 * r);
                bk1.u32[r] = *(const unsigned*)(krow + 32 + 2 * r);
            }
            v8f st = {};
            st = __builtin_amdgcn_wmma_f32_16x16x32_bf16(false, aq0.v, false, bk0.v,
                                                         (short)0, st, false, false);
            st = __builtin_amdgcn_wmma_f32_16x16x32_bf16(false, aq1.v, false, bk1.v,
                                                         (short)0, st, false, false);
            s[t] = st;
        }

        // bias + causal mask + online softmax (rows live across 16 lanes)
#pragma unroll
        for (int r = 0; r < 8; ++r) {
            int i  = i0 + (g << 3) + r;
            int ja = j0 + nloc, jb2 = j0 + 16 + nloc;
            float x0 = (ja  <= i) ? (s[0][r] + lutH[i - ja])  : -3.0e38f;
            float x1 = (jb2 <= i) ? (s[1][r] + lutH[i - jb2]) : -3.0e38f;
            float mx = fmaxf(x0, x1);
#pragma unroll
            for (int off = 8; off >= 1; off >>= 1) mx = fmaxf(mx, __shfl_xor(mx, off, 16));
            float mnew = fmaxf(m_run[r], mx);
            float corr = __expf(m_run[r] - mnew);
            float p0 = __expf(x0 - mnew);
            float p1 = __expf(x1 - mnew);
            float rs = p0 + p1;
#pragma unroll
            for (int off = 8; off >= 1; off >>= 1) rs += __shfl_xor(rs, off, 16);
            l_run[r] = l_run[r] * corr + rs;
            m_run[r] = mnew;
            o[0][r] *= corr; o[1][r] *= corr; o[2][r] *= corr; o[3][r] *= corr;
            int mrow = (g << 3) + r;
            smw[mrow * 32 + nloc]      = f2bf(p0);
            smw[mrow * 32 + 16 + nloc] = f2bf(p1);
        }
        asm volatile("s_wait_dscnt 0" ::: "memory");   // cross-lane LDS store->load (same wave)

        // reload P in A-layout (K-dim = 32 key positions)
        FragAB ap;
#pragma unroll
        for (int r = 0; r < 8; ++r) {
            int kk = (g << 4) + (r << 1);
            ap.u32[r] = *(const unsigned*)(smw + nloc * 32 + kk);
        }

        // O += P @ V  (V transposed: B-frag pairs contiguous along j)
        const unsigned short* vbase = Vt + ((size_t)(bh * DHEAD) << 11) + j0 + (g << 4);
#pragma unroll
        for (int db = 0; db < 4; ++db) {
            const unsigned short* vrow = vbase + ((size_t)(db * 16 + nloc) << 11);
            FragAB bv;
#pragma unroll
            for (int r = 0; r < 8; ++r) bv.u32[r] = *(const unsigned*)(vrow + 2 * r);
            o[db] = __builtin_amdgcn_wmma_f32_16x16x32_bf16(false, ap.v, false, bv.v,
                                                            (short)0, o[db], false, false);
        }
    }

    // epilogue: normalize by row-sum, store bf16 [b*n, h*d]
#pragma unroll
    for (int r = 0; r < 8; ++r) {
        float inv = 1.0f / l_run[r];
        int m = b * SEQ + i0 + (g << 3) + r;
        unsigned short* orow = O + (size_t)m * INNER + h * DHEAD + nloc;
        orow[0]  = f2bf(o[0][r] * inv);
        orow[16] = f2bf(o[1][r] * inv);
        orow[32] = f2bf(o[2][r] * inv);
        orow[48] = f2bf(o[3][r] * inv);
    }
}

// ---------------------------------------------------------------------------
extern "C" void kernel_launch(void* const* d_in, const int* in_sizes, int n_in,
                              void* d_out, int out_size, void* d_ws, size_t ws_size,
                              hipStream_t stream) {
    (void)in_sizes; (void)n_in; (void)out_size; (void)ws_size;
    const float* x   = (const float*)d_in[0];
    const float* Wq  = (const float*)d_in[1];
    const float* Wkv = (const float*)d_in[2];
    const float* Wo  = (const float*)d_in[3];
    const float* bo  = (const float*)d_in[4];
    const float* rel = (const float*)d_in[5];

    char* ws = (char*)d_ws;
    size_t off = 0;
    auto alloc = [&](size_t bytes) -> void* {
        void* p = ws + off;
        off = (off + bytes + 255) & ~(size_t)255;
        return p;
    };
    unsigned short* xb   = (unsigned short*)alloc((size_t)MROWS * DIM * 2);     // 8 MB
    unsigned short* wqT  = (unsigned short*)alloc((size_t)INNER * DIM * 2);     // 1 MB
    unsigned short* wkT  = (unsigned short*)alloc((size_t)INNER * DIM * 2);
    unsigned short* wvT  = (unsigned short*)alloc((size_t)INNER * DIM * 2);
    unsigned short* woT  = (unsigned short*)alloc((size_t)DIM * INNER * 2);
    unsigned short* qb   = (unsigned short*)alloc((size_t)MROWS * INNER * 2);   // 4 MB
    unsigned short* kbuf = (unsigned short*)alloc((size_t)MROWS * INNER * 2);
    unsigned short* vT   = (unsigned short*)alloc((size_t)BATCH * HEADS * DHEAD * SEQ * 2);
    unsigned short* attb = (unsigned short*)alloc((size_t)MROWS * INNER * 2);
    float*          lut  = (float*)alloc((size_t)HEADS * SEQ * 4);

    // conversions
    k_cvt_x<<<(MROWS * DIM + 255) / 256, 256, 0, stream>>>(x, xb, MROWS * DIM);
    k_cvt_wT<<<(DIM * INNER + 255) / 256, 256, 0, stream>>>(Wq,  wqT, DIM, INNER, INNER, 0);
    k_cvt_wT<<<(DIM * INNER + 255) / 256, 256, 0, stream>>>(Wkv, wkT, DIM, INNER, 2 * INNER, 0);
    k_cvt_wT<<<(DIM * INNER + 255) / 256, 256, 0, stream>>>(Wkv, wvT, DIM, INNER, 2 * INNER, INNER);
    k_cvt_wT<<<(INNER * DIM + 255) / 256, 256, 0, stream>>>(Wo,  woT, INNER, DIM, DIM, 0);
    k_bias_lut<<<(HEADS * SEQ + 255) / 256, 256, 0, stream>>>(rel, lut);

    // projections: M=4096, N=512, K=1024 -> 64 m-blocks x 8 strips = 512 blocks
    k_gemm_bf16<<<512, 128, 0, stream>>>(xb, wqT, qb,   MROWS, INNER, DIM, 0.125f, 0);
    k_gemm_bf16<<<512, 128, 0, stream>>>(xb, wkT, kbuf, MROWS, INNER, DIM, 1.0f,   0);
    k_gemm_bf16<<<512, 128, 0, stream>>>(xb, wvT, vT,   MROWS, INNER, DIM, 1.0f,   1);

    // attention: grid (query blocks of 64, b*h)
    k_flash<<<dim3(SEQ / 64, BATCH * HEADS), 128, 0, stream>>>(qb, kbuf, vT, lut, attb);

    // output projection: M=4096, N=1024, K=512 -> 64 m-blocks x 16 strips = 1024 blocks
    k_gemm_out<<<1024, 128, 0, stream>>>(attb, woT, bo, (float*)d_out, MROWS, DIM, INNER);
}
